// VTN_45440753992378
// MI455X (gfx1250) — compile-verified, matrix operand order checked by reading, not run on hardware
//
#include <hip/hip_runtime.h>
#include <hip/hip_bf16.h>
#include <math.h>

// ---------------- CDNA5 WMMA types ----------------
typedef __attribute__((ext_vector_type(16))) _Float16 v16h;
typedef __attribute__((ext_vector_type(8)))  float    v8f;

union FragH { v16h h; unsigned int u[8]; };

__device__ inline v8f v8f_zero() {
  v8f z;
#pragma unroll
  for (int e = 0; e < 8; e++) z[e] = 0.f;
  return z;
}

__device__ inline unsigned int cvt_pk(float a, float b) {
  union { _Float16 h[2]; unsigned int u; } p;
  p.h[0] = (_Float16)a;
  p.h[1] = (_Float16)b;
  return p.u;
}

__device__ inline unsigned int pk_hh(_Float16 a, _Float16 b) {
  union { _Float16 h[2]; unsigned int u; } p;
  p.h[0] = a;
  p.h[1] = b;
  return p.u;
}

__device__ inline float gelu_f(float x) {
  return 0.5f * x * (1.f + erff(x * 0.70710678118654752f));
}

// ============ weight convert + transpose: Wt[N][K] f16 <- W[K][N] f32 ======
// 64x64 tiles via LDS; grid (N/64, K/64), 256 threads.
__global__ __launch_bounds__(256) void convt_kernel(
    const float* __restrict__ W, _Float16* __restrict__ Wt, int Kk, int N) {
  __shared__ _Float16 t[64][65];
  const int bn = blockIdx.x * 64, bk = blockIdx.y * 64;
  for (int i = threadIdx.x; i < 64 * 64; i += 256) {
    int kk = i >> 6, nn = i & 63;  // coalesced along n
    t[nn][kk] = (_Float16)W[(unsigned)((bk + kk) * N + bn + nn)];
  }
  __syncthreads();
  for (int i = threadIdx.x; i < 64 * 32; i += 256) {
    int nn = i >> 5, kp = i & 31;  // coalesced along k, 2 f16 per store
    unsigned int u = pk_hh(t[nn][2 * kp], t[nn][2 * kp + 1]);
    *(unsigned int*)(Wt + (size_t)(bn + nn) * Kk + bk + 2 * kp) = u;
  }
}

// ================= Tiled WMMA GEMM (f16 in, f32 acc), pipelined ============
// C = act( A[M][K] f16 * Bt[N][K]^T f16 + bias )
// Block: 256 thr (8 waves, 2x4); block tile 64x256; wave tile 32x64;
// K-tile 64 (two 16x16x32 substeps). Register-staged software pipeline with
// UNCONDITIONAL clamped prefetch (no conditional live-ins -> no spills) and
// launch_bounds(256,1) to relax the VGPR occupancy target.
// K%64==0, M%64==0, N%256==0 for all uses here.
#define BM 64
#define BN 256
#define BK 64
#define PSTR 36  // padded pair stride (u32): 144B rows -> aligned b128

__global__ __launch_bounds__(256, 1) void gemm_wmma_kernel(
    const _Float16* __restrict__ A, int lda,
    const _Float16* __restrict__ Bt, int ldb,
    const float* __restrict__ bias,
    float* __restrict__ Cf, _Float16* __restrict__ Ch, int ldc,
    int N, int K, int act) {
  __shared__ unsigned int Alds[BM][PSTR];   // 9216 B
  __shared__ unsigned int BldsT[BN][PSTR];  // 36864 B (n-major)

  const int tid  = threadIdx.x;
  const int wave = tid >> 5;
  const int lane = tid & 31;
  const int mrow = lane & 15;
  const int half = lane >> 4;

  const int bm0 = blockIdx.y * BM;
  const int bn0 = blockIdx.x * BN;
  const int wm0 = (wave >> 2) * 32;  // 0 or 32
  const int wn0 = (wave & 3) * 64;   // 0..192

  // loop-invariant staging coordinates: chunk row base + 8-f16 chunk index
  const int sm = tid >> 3;           // 0..31 (A row / B row base, +j*32)
  const int sq = tid & 7;            // chunk within row
  const _Float16* Aptr = A + (size_t)(bm0 + sm) * lda + sq * 8;
  const _Float16* Bptr = Bt + (size_t)(bn0 + sm) * ldb + sq * 8;

  v8f acc[2][4];
#pragma unroll
  for (int i = 0; i < 2; i++)
#pragma unroll
    for (int j = 0; j < 4; j++) acc[i][j] = v8f_zero();

  // ---- prologue: prefetch tile 0 into registers ----
  uint4 pa[2], pb[8];
#pragma unroll
  for (int j = 0; j < 2; j++)
    pa[j] = *(const uint4*)(Aptr + (size_t)j * 32 * lda);
#pragma unroll
  for (int j = 0; j < 8; j++)
    pb[j] = *(const uint4*)(Bptr + (size_t)j * 32 * ldb);

  for (int k0 = 0; k0 < K; k0 += BK) {
    // ---- dump staged registers to LDS ----
#pragma unroll
    for (int j = 0; j < 2; j++)
      *(uint4*)&Alds[sm + j * 32][sq * 4] = pa[j];
#pragma unroll
    for (int j = 0; j < 8; j++)
      *(uint4*)&BldsT[sm + j * 32][sq * 4] = pb[j];
    __syncthreads();

    // ---- issue next tile's global loads unconditionally (clamped) ----
    // Last iteration harmlessly re-reads tile 0; values are never stored.
    const int kn = (k0 + BK < K) ? (k0 + BK) : 0;
#pragma unroll
    for (int j = 0; j < 2; j++)
      pa[j] = *(const uint4*)(Aptr + (size_t)j * 32 * lda + kn);
#pragma unroll
    for (int j = 0; j < 8; j++)
      pb[j] = *(const uint4*)(Bptr + (size_t)j * 32 * ldb + kn);

    // ---- compute: 2 substeps x (2x4) WMMA ----
#pragma unroll
    for (int kt = 0; kt < 2; kt++) {
      const int pbase = kt * 16 + half * 4;  // pair base, 16B aligned
      FragH fa[2], fb[4];
#pragma unroll
      for (int i = 0; i < 2; i++) {
        int m = wm0 + i * 16 + mrow;
        uint4 lo = *(const uint4*)&Alds[m][pbase];
        uint4 hi = *(const uint4*)&Alds[m][pbase + 8];
        fa[i].u[0] = lo.x; fa[i].u[1] = lo.y; fa[i].u[2] = lo.z; fa[i].u[3] = lo.w;
        fa[i].u[4] = hi.x; fa[i].u[5] = hi.y; fa[i].u[6] = hi.z; fa[i].u[7] = hi.w;
      }
#pragma unroll
      for (int j = 0; j < 4; j++) {
        int n = wn0 + j * 16 + mrow;
        uint4 lo = *(const uint4*)&BldsT[n][pbase];
        uint4 hi = *(const uint4*)&BldsT[n][pbase + 8];
        fb[j].u[0] = lo.x; fb[j].u[1] = lo.y; fb[j].u[2] = lo.z; fb[j].u[3] = lo.w;
        fb[j].u[4] = hi.x; fb[j].u[5] = hi.y; fb[j].u[6] = hi.z; fb[j].u[7] = hi.w;
      }
#pragma unroll
      for (int i = 0; i < 2; i++)
#pragma unroll
        for (int j = 0; j < 4; j++)
          acc[i][j] = __builtin_amdgcn_wmma_f32_16x16x32_f16(
              false, fa[i].h, false, fb[j].h, (short)0, acc[i][j], false, false);
    }
    __syncthreads();
  }

  // epilogue: lane owns column (lane&15); rows r + 8*half
#pragma unroll
  for (int i = 0; i < 2; i++)
#pragma unroll
    for (int j = 0; j < 4; j++) {
      int n = bn0 + wn0 + j * 16 + mrow;
      float bv = bias ? bias[n] : 0.f;
#pragma unroll
      for (int r = 0; r < 8; r++) {
        int m = bm0 + wm0 + i * 16 + r + (half << 3);
        float val = acc[i][j][r] + bv;
        if (act == 1) val = gelu_f(val);
        if (Cf) Cf[(size_t)m * ldc + n] = val;
        if (Ch) Ch[(size_t)m * ldc + n] = (_Float16)val;
      }
    }
}

// ================= Band attention (w=128, DH=64, H=12), f16 in/out ========
__global__ __launch_bounds__(32) void band_attn_kernel(
    const _Float16* __restrict__ Q, const _Float16* __restrict__ K,
    const _Float16* __restrict__ V, _Float16* __restrict__ O,
    int Bb, int S, int Dd) {
  __shared__ float    sc[16][384];
  __shared__ _Float16 pr[16][384];  // 768B rows -> aligned b128 reads

  const int lane = threadIdx.x;
  const int mrow = lane & 15;
  const int half = lane >> 4;
  const int nc = S >> 7;

  int idx = blockIdx.x;
  const int rb = idx & 7; idx >>= 3;
  const int c  = idx % nc; idx /= nc;
  const int hd = idx % 12;
  const int b  = idx / 12;

  const int qi_base = rb * 16;
  const int sq = c * 128 + qi_base + mrow;  // A-frag row (global)
  const _Float16* qrow = Q + (size_t)((b * S + sq) * Dd + hd * 64);
  const float scale = 0.125f;               // 1/sqrt(64)

  // ---- scores = scale * Q Kw^T, masked ----
  for (int nt = 0; nt < 24; nt++) {
    v8f acc = v8f_zero();
    const int j  = nt * 16 + mrow;          // key-window column (lane)
    const int kp = c * 128 + j - 128;       // global key position
    const bool okk = (kp >= 0) && (kp < S);
    const _Float16* krow = K + (size_t)((b * S + (okk ? kp : 0)) * Dd + hd * 64);
#pragma unroll
    for (int ks = 0; ks < 2; ks++) {
      const int pb = ks * 16 + half * 4;    // pair index base
      FragH fa, fb;
      {
        uint4 lo = *(const uint4*)(qrow + 2 * pb);
        uint4 hi = *(const uint4*)(qrow + 2 * (pb + 8));
        fa.u[0] = lo.x; fa.u[1] = lo.y; fa.u[2] = lo.z; fa.u[3] = lo.w;
        fa.u[4] = hi.x; fa.u[5] = hi.y; fa.u[6] = hi.z; fa.u[7] = hi.w;
      }
      if (okk) {
        uint4 lo = *(const uint4*)(krow + 2 * pb);
        uint4 hi = *(const uint4*)(krow + 2 * (pb + 8));
        fb.u[0] = lo.x; fb.u[1] = lo.y; fb.u[2] = lo.z; fb.u[3] = lo.w;
        fb.u[4] = hi.x; fb.u[5] = hi.y; fb.u[6] = hi.z; fb.u[7] = hi.w;
      } else {
#pragma unroll
        for (int v = 0; v < 8; v++) fb.u[v] = 0u;
      }
      acc = __builtin_amdgcn_wmma_f32_16x16x32_f16(
          false, fa.h, false, fb.h, (short)0, acc, false, false);
    }
#pragma unroll
    for (int r = 0; r < 8; r++) {
      int ml = r + (half << 3);
      int i  = qi_base + ml;                       // query index in chunk
      bool ok = okk && (j >= i) && (j <= i + 256); // |rel| <= w
      sc[ml][j] = ok ? acc[r] * scale : -1e9f;
    }
  }
  __syncthreads();

  // ---- per-row softmax: each half-wave covers 192 columns of its row ----
  {
    const int j0 = half * 192;
    float mx = -1e30f;
    for (int j = 0; j < 192; j++) mx = fmaxf(mx, sc[mrow][j0 + j]);
    mx = fmaxf(mx, __shfl_xor(mx, 16, 32));
    float sum = 0.f;
    for (int j = 0; j < 192; j++) {
      float e = expf(sc[mrow][j0 + j] - mx);
      sc[mrow][j0 + j] = e;
      sum += e;
    }
    sum += __shfl_xor(sum, 16, 32);
    float inv = 1.f / sum;
    for (int j = 0; j < 192; j++)
      pr[mrow][j0 + j] = (_Float16)(sc[mrow][j0 + j] * inv);
  }
  __syncthreads();

  // ---- out = P Vw ----
  for (int nt = 0; nt < 4; nt++) {
    v8f acc = v8f_zero();
    const int n = nt * 16 + mrow;           // dh column (lane)
#pragma unroll 4
    for (int ks = 0; ks < 12; ks++) {
      FragH fa, fb;
      {
        const int pb = ks * 16 + half * 4;
        uint4 lo = *(const uint4*)&pr[mrow][2 * pb];
        uint4 hi = *(const uint4*)&pr[mrow][2 * (pb + 8)];
        fa.u[0] = lo.x; fa.u[1] = lo.y; fa.u[2] = lo.z; fa.u[3] = lo.w;
        fa.u[4] = hi.x; fa.u[5] = hi.y; fa.u[6] = hi.z; fa.u[7] = hi.w;
      }
#pragma unroll
      for (int v = 0; v < 8; v++) {
        int kk = ks * 32 + ((v >> 2) << 4) + (half << 3) + ((v & 3) << 1);
        int kp0 = c * 128 + kk - 128;
        int kp1 = kp0 + 1;
        _Float16 h0 = (kp0 >= 0 && kp0 < S)
            ? V[(size_t)((b * S + kp0) * Dd + hd * 64 + n)] : (_Float16)0.f;
        _Float16 h1 = (kp1 >= 0 && kp1 < S)
            ? V[(size_t)((b * S + kp1) * Dd + hd * 64 + n)] : (_Float16)0.f;
        fb.u[v] = pk_hh(h0, h1);
      }
      acc = __builtin_amdgcn_wmma_f32_16x16x32_f16(
          false, fa.h, false, fb.h, (short)0, acc, false, false);
    }
#pragma unroll
    for (int r = 0; r < 8; r++) {
      int ml  = r + (half << 3);
      int sqr = c * 128 + qi_base + ml;
      O[(size_t)((b * S + sqr) * Dd + hd * 64 + n)] = (_Float16)acc[r];
    }
  }
}

// ====== residual + LayerNorm (D=768, one wave per row), f32 + f16 out =====
__global__ __launch_bounds__(32) void resln_kernel(
    float* __restrict__ h, _Float16* __restrict__ h16,
    const float* __restrict__ res,
    const float* __restrict__ gamma, const float* __restrict__ beta) {
  const int row = blockIdx.x;
  const int lane = threadIdx.x;
  const size_t base = (size_t)row * 768;
  float x[24];
  float s = 0.f;
#pragma unroll
  for (int i = 0; i < 6; i++) {
    int c0 = lane * 4 + i * 128;
    float4 v = *(const float4*)(h + base + c0);
    if (res) {
      float4 rv = *(const float4*)(res + base + c0);
      v.x += rv.x; v.y += rv.y; v.z += rv.z; v.w += rv.w;
    }
    x[4 * i + 0] = v.x; x[4 * i + 1] = v.y; x[4 * i + 2] = v.z; x[4 * i + 3] = v.w;
    s += v.x + v.y + v.z + v.w;
  }
  for (int o = 16; o > 0; o >>= 1) s += __shfl_xor(s, o, 32);
  float mean = s * (1.f / 768.f);
  float vs = 0.f;
#pragma unroll
  for (int i = 0; i < 24; i++) { float d = x[i] - mean; vs += d * d; }
  for (int o = 16; o > 0; o >>= 1) vs += __shfl_xor(vs, o, 32);
  float inv = rsqrtf(vs * (1.f / 768.f) + 1e-12f);
#pragma unroll
  for (int i = 0; i < 6; i++) {
    int c0 = lane * 4 + i * 128;
    float4 g = *(const float4*)(gamma + c0);
    float4 bb = *(const float4*)(beta + c0);
    float4 o;
    o.x = (x[4 * i + 0] - mean) * inv * g.x + bb.x;
    o.y = (x[4 * i + 1] - mean) * inv * g.y + bb.y;
    o.z = (x[4 * i + 2] - mean) * inv * g.z + bb.z;
    o.w = (x[4 * i + 3] - mean) * inv * g.w + bb.w;
    *(float4*)(h + base + c0) = o;
    if (h16) {
      uint2 uu;
      uu.x = cvt_pk(o.x, o.y);
      uu.y = cvt_pk(o.z, o.w);
      *(uint2*)(h16 + base + c0) = uu;
    }
  }
}

// ================= patch rearrange (writes f16 GEMM input) =================
__global__ void rearrange_kernel(const float* __restrict__ x,
                                 _Float16* __restrict__ xr, int total) {
  int i = blockIdx.x * blockDim.x + threadIdx.x;
  if (i >= total) return;
  int f = i % 192;  int rest = i / 192;
  int s = rest % 1024; int nn = rest / 1024;
  int cc = f % 3;  int cr3 = f / 3;
  int i4 = cr3 & 3; int i2 = (cr3 >> 2) & 3; int i0 = cr3 >> 4;
  int i5 = s & 7;  int i3 = (s >> 3) & 7;   int i1 = s >> 6;
  int t = i0 * 16 + i1, wd = i2 * 8 + i3, hh = i4 * 8 + i5;
  xr[i] = (_Float16)x[(unsigned)((((nn * 3 + cc) * 64 + t) * 32 + wd) * 32 + hh)];
}

// ================= embeddings add =================
__global__ void add_emb_kernel(float* __restrict__ h, const float* __restrict__ pos,
                               const float* __restrict__ tok, int S, int Dd, int total) {
  int i = blockIdx.x * blockDim.x + threadIdx.x;
  if (i >= total) return;
  int d = i % Dd;
  int s = (i / Dd) % S;
  h[i] += pos[(unsigned)(s * Dd + d)] + tok[d];
}

// ================= small GEMM for pooled head (M=8, f32) =================
// act: 0=none, 1=tanh, 2=gelu
__global__ void small_gemm_kernel(const float* __restrict__ A, int arowstride,
                                  const float* __restrict__ W, int ldw,
                                  const float* __restrict__ bias,
                                  float* __restrict__ C, int ldc,
                                  int M, int N, int Kk, int act) {
  int i = blockIdx.x * blockDim.x + threadIdx.x;
  if (i >= M * N) return;
  int m = i / N, n = i % N;
  const float* a = A + (size_t)m * arowstride;
  float s = bias ? bias[n] : 0.f;
  for (int k = 0; k < Kk; k++) s += a[k] * W[(unsigned)(k * ldw + n)];
  if (act == 1) s = tanhf(s);
  else if (act == 2) s = gelu_f(s);
  C[(size_t)m * ldc + n] = s;
}

// ======================= host orchestration =======================
extern "C" void kernel_launch(void* const* d_in, const int* in_sizes, int n_in,
                              void* d_out, int out_size, void* d_ws, size_t ws_size,
                              hipStream_t stream) {
  const float* x        = (const float*)d_in[0];
  // d_in[1] = position_ids (unused by reference forward)
  const float* proj_w   = (const float*)d_in[2];
  const float* proj_b   = (const float*)d_in[3];
  const float* pos_emb  = (const float*)d_in[4];
  const float* tok_emb  = (const float*)d_in[5];
  const float* emb_ln_s = (const float*)d_in[6];
  const float* emb_ln_b = (const float*)d_in[7];
  const float* wq = (const float*)d_in[8];   const float* bq = (const float*)d_in[9];
  const float* wk = (const float*)d_in[10];  const float* bk = (const float*)d_in[11];
  const float* wv = (const float*)d_in[12];  const float* bv = (const float*)d_in[13];
  const float* wo = (const float*)d_in[14];  const float* bo = (const float*)d_in[15];
  const float* ln1_s = (const float*)d_in[16]; const float* ln1_b = (const float*)d_in[17];
  const float* wi = (const float*)d_in[18];  const float* bi = (const float*)d_in[19];
  const float* wo2 = (const float*)d_in[20]; const float* bo2 = (const float*)d_in[21];
  const float* ln2_s = (const float*)d_in[22]; const float* ln2_b = (const float*)d_in[23];
  const float* pool_w = (const float*)d_in[24]; const float* pool_b = (const float*)d_in[25];
  const float* head_ln_s = (const float*)d_in[26]; const float* head_ln_b = (const float*)d_in[27];
  const float* head_w1 = (const float*)d_in[28]; const float* head_b1 = (const float*)d_in[29];
  const float* head_w2 = (const float*)d_in[30]; const float* head_b2 = (const float*)d_in[31];

  const int Bx = 8, S = 1024, D = 768, Ii = 3072, Hh = 12, Ll = 3;
  const int BS = Bx * S;  // 8192 rows
  const size_t nHS = (size_t)BS * D;

  // ---- workspace layout (16B-aligned chunks) ----
  char* w = (char*)d_ws;
  auto af32 = [&](size_t n) { float* p = (float*)w; w += ((n * 4 + 15) & ~(size_t)15); return p; };
  auto af16 = [&](size_t n) { _Float16* p = (_Float16*)w; w += ((n * 2 + 15) & ~(size_t)15); return p; };

  float*     Hf   = af32(nHS);            // residual stream (f32 master)
  float*     AW   = af32(nHS);            // pre-residual GEMM output (f32)
  _Float16*  H16  = af16(nHS);            // f16 shadow of Hf
  _Float16*  Q16  = af16(nHS);
  _Float16*  K16  = af16(nHS);
  _Float16*  V16  = af16(nHS);
  _Float16*  A16  = af16(nHS);            // attention output
  _Float16*  G16  = af16((size_t)BS * Ii);
  _Float16*  XR16 = af16((size_t)BS * 192);
  float*     POOLED = af32((size_t)Bx * D);
  float*     Zb     = af32((size_t)Bx * 512);
  _Float16*  projWt = af16((size_t)D * 192);        // [768][192]
  _Float16*  wqT = af16((size_t)Ll * D * D);        // [768][768] per layer
  _Float16*  wkT = af16((size_t)Ll * D * D);
  _Float16*  wvT = af16((size_t)Ll * D * D);
  _Float16*  woT = af16((size_t)Ll * D * D);
  _Float16*  wiT = af16((size_t)Ll * D * Ii);       // [3072][768] per layer
  _Float16*  wo2T = af16((size_t)Ll * D * Ii);      // [768][3072] per layer

  // ---- 0. weight convert + transpose (f32 [K][N] -> f16 [N][K]) ----
  convt_kernel<<<dim3(D / 64, 192 / 64), 256, 0, stream>>>(proj_w, projWt, 192, D);
  for (int l = 0; l < Ll; l++) {
    const size_t oDD = (size_t)l * D * D, oDI = (size_t)l * D * Ii;
    dim3 gdd(D / 64, D / 64);
    convt_kernel<<<gdd, 256, 0, stream>>>(wq + oDD, wqT + oDD, D, D);
    convt_kernel<<<gdd, 256, 0, stream>>>(wk + oDD, wkT + oDD, D, D);
    convt_kernel<<<gdd, 256, 0, stream>>>(wv + oDD, wvT + oDD, D, D);
    convt_kernel<<<gdd, 256, 0, stream>>>(wo + oDD, woT + oDD, D, D);
    convt_kernel<<<dim3(Ii / 64, D / 64), 256, 0, stream>>>(wi + oDI, wiT + oDI, D, Ii);
    convt_kernel<<<dim3(D / 64, Ii / 64), 256, 0, stream>>>(wo2 + oDI, wo2T + oDI, Ii, D);
  }

  // ---- 1. patch rearrange (f16) ----
  {
    int total = BS * 192;
    rearrange_kernel<<<(total + 255) / 256, 256, 0, stream>>>(x, XR16, total);
  }
  // ---- 2. projection GEMM: (BS,192) x (192,768) -> f32 Hf ----
  {
    dim3 g(D / BN, BS / BM);
    gemm_wmma_kernel<<<g, 256, 0, stream>>>(XR16, 192, projWt, 192, proj_b,
                                            Hf, nullptr, D, D, 192, 0);
  }
  // ---- 3. embeddings + LN (writes Hf + H16) ----
  {
    int total = BS * D;
    add_emb_kernel<<<(total + 255) / 256, 256, 0, stream>>>(Hf, pos_emb, tok_emb, S, D, total);
    resln_kernel<<<BS, 32, 0, stream>>>(Hf, H16, nullptr, emb_ln_s, emb_ln_b);
  }
  // ---- 4. transformer layers ----
  dim3 gDD(D / BN, BS / BM);
  dim3 gDI(Ii / BN, BS / BM);
  const int attn_blocks = Bx * Hh * (S / 128) * 8;
  for (int l = 0; l < Ll; l++) {
    const size_t oDD = (size_t)l * D * D, oD = (size_t)l * D;
    const size_t oDI = (size_t)l * D * Ii, oI = (size_t)l * Ii;
    gemm_wmma_kernel<<<gDD, 256, 0, stream>>>(H16, D, wqT + oDD, D, bq + oD,
                                              nullptr, Q16, D, D, D, 0);
    gemm_wmma_kernel<<<gDD, 256, 0, stream>>>(H16, D, wkT + oDD, D, bk + oD,
                                              nullptr, K16, D, D, D, 0);
    gemm_wmma_kernel<<<gDD, 256, 0, stream>>>(H16, D, wvT + oDD, D, bv + oD,
                                              nullptr, V16, D, D, D, 0);
    band_attn_kernel<<<attn_blocks, 32, 0, stream>>>(Q16, K16, V16, A16, Bx, S, D);
    gemm_wmma_kernel<<<gDD, 256, 0, stream>>>(A16, D, woT + oDD, D, bo + oD,
                                              AW, nullptr, D, D, D, 0);
    resln_kernel<<<BS, 32, 0, stream>>>(Hf, H16, AW, ln1_s + oD, ln1_b + oD);
    gemm_wmma_kernel<<<gDI, 256, 0, stream>>>(H16, D, wiT + oDI, D, bi + oI,
                                              nullptr, G16, Ii, Ii, D, 1);
    gemm_wmma_kernel<<<gDD, 256, 0, stream>>>(G16, Ii, wo2T + oDI, Ii, bo2 + oD,
                                              AW, nullptr, D, D, Ii, 0);
    resln_kernel<<<BS, 32, 0, stream>>>(Hf, H16, AW, ln2_s + oD, ln2_b + oD);
  }
  // ---- 5. pooled head (f32): tanh -> LN -> gelu -> logits ----
  small_gemm_kernel<<<(Bx * D + 255) / 256, 256, 0, stream>>>(
      Hf, S * D, pool_w, D, pool_b, POOLED, D, Bx, D, D, 1);
  resln_kernel<<<Bx, 32, 0, stream>>>(POOLED, nullptr, nullptr, head_ln_s, head_ln_b);
  small_gemm_kernel<<<(Bx * 512 + 255) / 256, 256, 0, stream>>>(
      POOLED, D, head_w1, 512, head_b1, Zb, 512, Bx, 512, D, 2);
  small_gemm_kernel<<<1, 256, 0, stream>>>(
      Zb, 512, head_w2, 3, head_b2, (float*)d_out, 3, Bx, 3, 512, 0);
}
